// LlamaAttention_46162308498114
// MI455X (gfx1250) — matrix operand order repaired and checked by reading, not compiled, not run on hardware
//
#include <hip/hip_runtime.h>

#define T_SEQ  2048
#define HID    4096
#define NQH    32
#define NKVH   8
#define HD     128
#define QDIM   (NQH*HD)    // 4096
#define KVDIM  (NKVH*HD)   // 1024

typedef unsigned short u16;
typedef unsigned long long u64;
typedef __attribute__((ext_vector_type(4)))  unsigned int u32x4;
typedef __attribute__((ext_vector_type(4)))  int          i32x4;
typedef __attribute__((ext_vector_type(8)))  int          i32x8;
typedef __attribute__((ext_vector_type(4)))  float        f32x4;
typedef __attribute__((ext_vector_type(16))) __bf16       v16bf;
typedef __attribute__((ext_vector_type(8)))  float        v8f;

union Frag  { u32x4 u[2]; v16bf v; };
union Row16 { u32x4 v[2]; u16 e[16]; };

__device__ inline u16 f2bf(float f) { __bf16 h = (__bf16)f; return __builtin_bit_cast(u16, h); }
__device__ inline float bf2f(u16 b) { __bf16 h = __builtin_bit_cast(__bf16, b); return (float)h; }

__device__ inline v8f wmma_bf16(v16bf a, v16bf b, v8f c) {
  return __builtin_amdgcn_wmma_f32_16x16x32_bf16(false, a, false, b, (short)0, c, false, false);
}

__device__ inline void store_out(float* p, float v) { *p = v; }
__device__ inline void store_out(u16*   p, float v) { *p = f2bf(v); }

// TDM: async tensor load global -> LDS. Arity differs between toolchains.
__device__ inline void tdm_load(u32x4 g0, i32x8 g1) {
  i32x4 z4 = (i32x4)0;
#if __clang_major__ >= 23
  i32x8 z8 = (i32x8)0;
  __builtin_amdgcn_tensor_load_to_lds(g0, g1, z4, z4, z8, 0);
#else
  __builtin_amdgcn_tensor_load_to_lds(g0, g1, z4, z4, 0);
#endif
}

// D# for a 2D bf16 tile: tile_d0 x tile_d1 elements, row stride stride0 elements.
__device__ inline void tdm_load_tile_2d(u64 ga, unsigned lds,
                                        int tile_d0, int tile_d1, int stride0) {
  u32x4 g0;
  g0[0] = 1u;                                    // count=1, user descriptor
  g0[1] = lds;                                   // lds_addr (bytes)
  g0[2] = (unsigned)ga;                          // global_addr[31:0]
  g0[3] = (unsigned)(ga >> 32) | 0x80000000u;    // global_addr[56:32] | type=2
  i32x8 g1;
  g1[0] = 0x00010000;                            // data_size=1 (2 bytes)
  g1[1] = (tile_d0 << 16);                       // tensor_dim0[15:0]
  g1[2] = (tile_d1 << 16);                       // td0 hi | tensor_dim1[15:0]
  g1[3] = (tile_d0 << 16);                       // td1 hi | tile_dim0
  g1[4] = tile_d1;                               // tile_dim1 | tile_dim2=0
  g1[5] = stride0;                               // tensor_dim0_stride[31:0]
  g1[6] = 0;
  g1[7] = 0;
  tdm_load(g0, g1);
}

__device__ inline unsigned lds_off(const void* p) {
  return (unsigned)(u64)p;                       // low 32 bits = LDS byte offset
}

// ---------------------------------------------------------------- f32 -> bf16
__global__ void cvt_bf16_kernel(const float* __restrict__ in, u16* __restrict__ out, size_t n) {
  size_t i = ((size_t)blockIdx.x * blockDim.x + threadIdx.x) * 4;
  size_t stride = (size_t)gridDim.x * blockDim.x * 4;
  for (; i < n; i += stride) {
    f32x4 v = *(const f32x4*)(in + i);
    *(unsigned*)(out + i)     = (unsigned)f2bf(v[0]) | ((unsigned)f2bf(v[1]) << 16);
    *(unsigned*)(out + i + 2) = (unsigned)f2bf(v[2]) | ((unsigned)f2bf(v[3]) << 16);
  }
}

// ---------------------------------------------------------------- bf16 GEMM
// C[M,N] = A[M,K]*B[K,N]; tile 128x128, BK=64; TDM double-buffered pipeline.
// Dynamic LDS: As[2][128][64] | Bsh[2][64][128] | Bts[128][64]  = 80 KB.
template <typename OutT>
__global__ void __launch_bounds__(256) gemm_bf16_kernel(
    const u16* __restrict__ A, const u16* __restrict__ B, OutT* __restrict__ C,
    int M, int N, int K) {
  extern __shared__ __align__(16) u16 gsm[];
  u16 (*As)[128][64]  = (u16(*)[128][64])gsm;                   // 2 buffers
  u16 (*Bsh)[64][128] = (u16(*)[64][128])(gsm + 2 * 128 * 64);  // 2 buffers
  u16 (*Bts)[64]      = (u16(*)[64])(gsm + 4 * 128 * 64);       // [128][64]

  const int tid  = threadIdx.x;
  const int lane = tid & 31;
  const int wave = tid >> 5;
  const int wm   = wave >> 1;
  const int wn   = wave & 1;
  const int hlf  = lane >> 4;
  const int lr   = lane & 15;
  const int row0 = blockIdx.y * 128;
  const int col0 = blockIdx.x * 128;

  v8f acc[2][4];
#pragma unroll
  for (int mt = 0; mt < 2; ++mt)
#pragma unroll
    for (int nt = 0; nt < 4; ++nt)
#pragma unroll
      for (int r = 0; r < 8; ++r) acc[mt][nt][r] = 0.0f;

  const int nit = K >> 6;
  if (wave == 0) {  // prologue: tile 0
    tdm_load_tile_2d((u64)(A + (size_t)row0 * K), lds_off(&As[0][0][0]), 64, 128, K);
    tdm_load_tile_2d((u64)(B + col0),             lds_off(&Bsh[0][0][0]), 128, 64, N);
  }

  for (int it = 0; it < nit; ++it) {
    const int buf = it & 1;
    if (wave == 0) {
      if (it + 1 < nit) {  // prefetch next tile into other buffer
        tdm_load_tile_2d((u64)(A + (size_t)row0 * K + (it + 1) * 64),
                         lds_off(&As[buf ^ 1][0][0]), 64, 128, K);
        tdm_load_tile_2d((u64)(B + (size_t)(it + 1) * 64 * N + col0),
                         lds_off(&Bsh[buf ^ 1][0][0]), 128, 64, N);
        __builtin_amdgcn_s_wait_tensorcnt(2);   // current tile's 2 DMAs done
      } else {
        __builtin_amdgcn_s_wait_tensorcnt(0);
      }
    }
    __syncthreads();

    { // transpose B tile: pack k-pairs -> b32 stores
      int kk = (tid >> 3) * 2, bn = (tid & 7) * 16;
      Row16 r0, r1;
      r0.v[0] = *(const u32x4*)&Bsh[buf][kk][bn];
      r0.v[1] = *(const u32x4*)&Bsh[buf][kk][bn + 8];
      r1.v[0] = *(const u32x4*)&Bsh[buf][kk + 1][bn];
      r1.v[1] = *(const u32x4*)&Bsh[buf][kk + 1][bn + 8];
#pragma unroll
      for (int i = 0; i < 16; ++i)
        *(unsigned*)&Bts[bn + i][kk] = (unsigned)r0.e[i] | ((unsigned)r1.e[i] << 16);
    }
    __syncthreads();

#pragma unroll
    for (int kd = 0; kd < 2; ++kd) {
      Frag a[2], b[4];
#pragma unroll
      for (int mt = 0; mt < 2; ++mt) {
        int r = wm * 32 + mt * 16 + lr;
        a[mt].u[0] = *(const u32x4*)&As[buf][r][kd * 32 + hlf * 8];
        a[mt].u[1] = *(const u32x4*)&As[buf][r][kd * 32 + 16 + hlf * 8];
      }
#pragma unroll
      for (int nt = 0; nt < 4; ++nt) {
        int n = wn * 64 + nt * 16 + lr;
        b[nt].u[0] = *(const u32x4*)&Bts[n][kd * 32 + hlf * 8];
        b[nt].u[1] = *(const u32x4*)&Bts[n][kd * 32 + 16 + hlf * 8];
      }
#pragma unroll
      for (int mt = 0; mt < 2; ++mt)
#pragma unroll
        for (int nt = 0; nt < 4; ++nt)
          acc[mt][nt] = wmma_bf16(a[mt].v, b[nt].v, acc[mt][nt]);
    }
    __syncthreads();   // gate buffer reuse by next TDM issue
  }

#pragma unroll
  for (int mt = 0; mt < 2; ++mt)
#pragma unroll
    for (int nt = 0; nt < 4; ++nt) {
      int rbase = row0 + wm * 32 + mt * 16 + hlf * 8;
      int c     = col0 + wn * 64 + nt * 16 + lr;
      OutT* p = C + (size_t)rbase * N + c;
#pragma unroll
      for (int r = 0; r < 8; ++r) { store_out(p, acc[mt][nt][r]); p += N; }
    }
}

#define GEMM_LDS_BYTES ((2*128*64 + 2*64*128 + 128*64) * sizeof(u16))  // 80 KB

// ---------------------------------------------------------------- NeoX RoPE
__global__ void rope_kernel(const u16* __restrict__ in, u16* __restrict__ out,
                            const int* __restrict__ positions, int nheads, float scale) {
  int t = blockIdx.x, h = blockIdx.y, j = threadIdx.x;  // 64 threads
  float pos = (float)positions[t];
  float inv = __powf(10000.0f, -(float)j / 64.0f);
  float s, c;
  __sincosf(pos * inv, &s, &c);
  const u16* p = in  + ((size_t)t * nheads + h) * HD;
  u16*       o = out + ((size_t)t * nheads + h) * HD;
  float x1 = bf2f(p[j]), x2 = bf2f(p[j + 64]);
  o[j]      = f2bf((x1 * c - x2 * s) * scale);
  o[j + 64] = f2bf((x2 * c + x1 * s) * scale);
}

// ---------------------------------------------------------------- flash attention
// Q [T][QDIM] (rope'd, pre-scaled), K/V [T][KVDIM], O [T][QDIM]; all bf16.
// 8 waves, 128 q-rows/block, 64-key tiles; TDM double-buffered K/V pipeline.
// Dynamic LDS: Ksh[2][64][HD] | Vsh[2][64][HD] | Vts[HD][64] | Psh[8][16][64] = 96 KB.
#define NEG_BIG (-3.0e38f)
__global__ void __launch_bounds__(256) flash_attn_kernel(
    const u16* __restrict__ Q, const u16* __restrict__ Kc, const u16* __restrict__ Vc,
    u16* __restrict__ O) {
  extern __shared__ __align__(16) u16 fsm[];
  u16 (*Ksh)[64][HD] = (u16(*)[64][HD])fsm;                        // 2 buffers
  u16 (*Vsh)[64][HD] = (u16(*)[64][HD])(fsm + 2 * 64 * HD);        // 2 buffers
  u16 (*Vts)[64]     = (u16(*)[64])(fsm + 4 * 64 * HD);            // [HD][64]
  u16 (*Psh)[16][64] = (u16(*)[16][64])(fsm + 4 * 64 * HD + HD * 64);

  const int h    = blockIdx.y;
  const int kvh  = h >> 2;
  const int q0   = blockIdx.x * 128;
  const int tid  = threadIdx.x;
  const int lane = tid & 31;
  const int wave = tid >> 5;
  const int hlf  = lane >> 4;
  const int lr   = lane & 15;

  const int nit = (q0 + 128) >> 6;   // causal: # of 64-key tiles
  if (wave == 0) {  // prologue: tile 0
    tdm_load_tile_2d((u64)(Kc + kvh * HD), lds_off(&Ksh[0][0][0]), HD, 64, KVDIM);
    tdm_load_tile_2d((u64)(Vc + kvh * HD), lds_off(&Vsh[0][0][0]), HD, 64, KVDIM);
  }

  // Q fragments (A operand: row = lane%16, half selects K-chunk) — overlaps TDM
  Frag aq[4];
  {
    const u16* qptr = Q + (size_t)(q0 + wave * 16 + lr) * QDIM + h * HD;
#pragma unroll
    for (int kd = 0; kd < 4; ++kd) {
      aq[kd].u[0] = *(const u32x4*)(qptr + kd * 32 + hlf * 8);
      aq[kd].u[1] = *(const u32x4*)(qptr + kd * 32 + 16 + hlf * 8);
    }
  }

  v8f oacc[8];
  float m[8], l[8];
#pragma unroll
  for (int dt = 0; dt < 8; ++dt)
#pragma unroll
    for (int r = 0; r < 8; ++r) oacc[dt][r] = 0.0f;
#pragma unroll
  for (int r = 0; r < 8; ++r) { m[r] = NEG_BIG; l[r] = 0.0f; }

  for (int it = 0; it < nit; ++it) {
    const int k0  = it * 64;
    const int buf = it & 1;
    if (wave == 0) {
      if (it + 1 < nit) {
        tdm_load_tile_2d((u64)(Kc + (size_t)(k0 + 64) * KVDIM + kvh * HD),
                         lds_off(&Ksh[buf ^ 1][0][0]), HD, 64, KVDIM);
        tdm_load_tile_2d((u64)(Vc + (size_t)(k0 + 64) * KVDIM + kvh * HD),
                         lds_off(&Vsh[buf ^ 1][0][0]), HD, 64, KVDIM);
        __builtin_amdgcn_s_wait_tensorcnt(2);
      } else {
        __builtin_amdgcn_s_wait_tensorcnt(0);
      }
    }
    __syncthreads();

    { // transpose V tile: pack key pairs -> b32 stores
      int kk = (tid >> 3) * 2, c = (tid & 7) * 16;
      Row16 r0, r1;
      r0.v[0] = *(const u32x4*)&Vsh[buf][kk][c];
      r0.v[1] = *(const u32x4*)&Vsh[buf][kk][c + 8];
      r1.v[0] = *(const u32x4*)&Vsh[buf][kk + 1][c];
      r1.v[1] = *(const u32x4*)&Vsh[buf][kk + 1][c + 8];
#pragma unroll
      for (int i = 0; i < 16; ++i)
        *(unsigned*)&Vts[c + i][kk] = (unsigned)r0.e[i] | ((unsigned)r1.e[i] << 16);
    }
    __syncthreads();

    // S = Q @ K^T
    v8f sacc[4];
#pragma unroll
    for (int nt = 0; nt < 4; ++nt)
#pragma unroll
      for (int r = 0; r < 8; ++r) sacc[nt][r] = 0.0f;
#pragma unroll
    for (int nt = 0; nt < 4; ++nt) {
      int key = nt * 16 + lr;
#pragma unroll
      for (int kd = 0; kd < 4; ++kd) {
        Frag bk;
        bk.u[0] = *(const u32x4*)&Ksh[buf][key][kd * 32 + hlf * 8];
        bk.u[1] = *(const u32x4*)&Ksh[buf][key][kd * 32 + 16 + hlf * 8];
        sacc[nt] = wmma_bf16(aq[kd].v, bk.v, sacc[nt]);
      }
    }

    // causal mask + row max (C layout: row = r + 8*hlf)
    float mnew[8];
#pragma unroll
    for (int r = 0; r < 8; ++r) mnew[r] = m[r];
#pragma unroll
    for (int nt = 0; nt < 4; ++nt) {
      int key = k0 + nt * 16 + lr;
#pragma unroll
      for (int r = 0; r < 8; ++r) {
        int qr = q0 + wave * 16 + hlf * 8 + r;
        float s = (key <= qr) ? sacc[nt][r] : NEG_BIG;
        sacc[nt][r] = s;
        mnew[r] = fmaxf(mnew[r], s);
      }
    }
#pragma unroll
    for (int r = 0; r < 8; ++r) {
      mnew[r] = fmaxf(mnew[r], __shfl_xor(mnew[r], 1, 32));
      mnew[r] = fmaxf(mnew[r], __shfl_xor(mnew[r], 2, 32));
      mnew[r] = fmaxf(mnew[r], __shfl_xor(mnew[r], 4, 32));
      mnew[r] = fmaxf(mnew[r], __shfl_xor(mnew[r], 8, 32));
    }

    float rowsum[8];
#pragma unroll
    for (int r = 0; r < 8; ++r) rowsum[r] = 0.0f;
#pragma unroll
    for (int nt = 0; nt < 4; ++nt)
#pragma unroll
      for (int r = 0; r < 8; ++r) {
        float p = __expf(sacc[nt][r] - mnew[r]);
        rowsum[r] += p;
        Psh[wave][hlf * 8 + r][nt * 16 + lr] = f2bf(p);
      }
#pragma unroll
    for (int r = 0; r < 8; ++r) {
      rowsum[r] += __shfl_xor(rowsum[r], 1, 32);
      rowsum[r] += __shfl_xor(rowsum[r], 2, 32);
      rowsum[r] += __shfl_xor(rowsum[r], 4, 32);
      rowsum[r] += __shfl_xor(rowsum[r], 8, 32);
    }
    float sc[8];
#pragma unroll
    for (int r = 0; r < 8; ++r) {
      sc[r] = __expf(m[r] - mnew[r]);
      l[r] = l[r] * sc[r] + rowsum[r];
      m[r] = mnew[r];
    }
#pragma unroll
    for (int dt = 0; dt < 8; ++dt)
#pragma unroll
      for (int r = 0; r < 8; ++r) oacc[dt][r] *= sc[r];

    // O += P @ V (same-wave LDS store->load: DS ops in-order per wave)
#pragma unroll
    for (int kd2 = 0; kd2 < 2; ++kd2) {
      Frag ap;
      ap.u[0] = *(const u32x4*)&Psh[wave][lr][kd2 * 32 + hlf * 8];
      ap.u[1] = *(const u32x4*)&Psh[wave][lr][kd2 * 32 + 16 + hlf * 8];
#pragma unroll
      for (int dt = 0; dt < 8; ++dt) {
        Frag bv;
        int d = dt * 16 + lr;
        bv.u[0] = *(const u32x4*)&Vts[d][kd2 * 32 + hlf * 8];
        bv.u[1] = *(const u32x4*)&Vts[d][kd2 * 32 + 16 + hlf * 8];
        oacc[dt] = wmma_bf16(ap.v, bv.v, oacc[dt]);
      }
    }
    __syncthreads();   // gate buffer reuse by next TDM issue
  }

  float invl[8];
#pragma unroll
  for (int r = 0; r < 8; ++r) invl[r] = 1.0f / l[r];
#pragma unroll
  for (int dt = 0; dt < 8; ++dt)
#pragma unroll
    for (int r = 0; r < 8; ++r) {
      int row = q0 + wave * 16 + hlf * 8 + r;
      int col = h * HD + dt * 16 + lr;
      O[(size_t)row * QDIM + col] = f2bf(oacc[dt][r] * invl[r]);
    }
}

#define FLASH_LDS_BYTES ((2*64*HD + 2*64*HD + HD*64 + 8*16*64) * sizeof(u16))  // 96 KB

// ---------------------------------------------------------------- launcher
extern "C" void kernel_launch(void* const* d_in, const int* in_sizes, int n_in,
                              void* d_out, int out_size, void* d_ws, size_t ws_size,
                              hipStream_t stream) {
  const int*   positions = (const int*)d_in[0];
  const float* hidden    = (const float*)d_in[1];
  const float* wq        = (const float*)d_in[2];
  const float* wk        = (const float*)d_in[3];
  const float* wv        = (const float*)d_in[4];
  const float* wo        = (const float*)d_in[5];
  float* out = (float*)d_out;
  char*  w   = (char*)d_ws;

  const size_t MB = 1024ull * 1024ull;
  u16* Xbf  = (u16*)(w + 0);          // 16 MB
  u16* Wqb  = (u16*)(w + 16 * MB);    // 32 MB
  u16* Wkb  = (u16*)(w + 48 * MB);    //  8 MB
  u16* Wvb  = (u16*)(w + 56 * MB);    //  8 MB
  u16* Wob  = (u16*)(w + 64 * MB);    // 32 MB
  u16* Qraw = (u16*)(w + 96 * MB);    // 16 MB
  u16* Kraw = (u16*)(w + 112 * MB);   //  4 MB
  u16* Vbf  = (u16*)(w + 116 * MB);   //  4 MB
  u16* Qr   = (u16*)(w + 120 * MB);   // 16 MB
  u16* Kr   = (u16*)(w + 136 * MB);   //  4 MB
  u16* Attn = (u16*)(w + 140 * MB);   // 16 MB  (total 156 MB)

  cvt_bf16_kernel<<<1024, 256, 0, stream>>>(hidden, Xbf, (size_t)T_SEQ * HID);
  cvt_bf16_kernel<<<2048, 256, 0, stream>>>(wq, Wqb, (size_t)HID * QDIM);
  cvt_bf16_kernel<<<512, 256, 0, stream>>>(wk, Wkb, (size_t)HID * KVDIM);
  cvt_bf16_kernel<<<512, 256, 0, stream>>>(wv, Wvb, (size_t)HID * KVDIM);
  cvt_bf16_kernel<<<2048, 256, 0, stream>>>(wo, Wob, (size_t)QDIM * HID);

  gemm_bf16_kernel<<<dim3(QDIM / 128, T_SEQ / 128), 256, GEMM_LDS_BYTES, stream>>>(
      Xbf, Wqb, Qraw, T_SEQ, QDIM, HID);
  gemm_bf16_kernel<<<dim3(KVDIM / 128, T_SEQ / 128), 256, GEMM_LDS_BYTES, stream>>>(
      Xbf, Wkb, Kraw, T_SEQ, KVDIM, HID);
  gemm_bf16_kernel<<<dim3(KVDIM / 128, T_SEQ / 128), 256, GEMM_LDS_BYTES, stream>>>(
      Xbf, Wvb, Vbf, T_SEQ, KVDIM, HID);

  rope_kernel<<<dim3(T_SEQ, NQH), 64, 0, stream>>>(Qraw, Qr, positions, NQH, 0.08838834764831845f);
  rope_kernel<<<dim3(T_SEQ, NKVH), 64, 0, stream>>>(Kraw, Kr, positions, NKVH, 1.0f);

  flash_attn_kernel<<<dim3(T_SEQ / 128, NQH), 256, FLASH_LDS_BYTES, stream>>>(Qr, Kr, Vbf, Attn);

  gemm_bf16_kernel<<<dim3(HID / 128, T_SEQ / 128), 256, GEMM_LDS_BYTES, stream>>>(
      Attn, Wob, out, T_SEQ, HID, QDIM);
}
